// TransducerDecoder_82257213653600
// MI455X (gfx1250) — compile-verified, matrix operand order checked by reading, not run on hardware
//
#include <hip/hip_runtime.h>
#include <cstddef>

// Problem constants (from reference)
constexpr int kB  = 4;
constexpr int kT  = 384;
constexpr int kU  = 96;
constexpr int kU1 = 97;     // U+1 (blank prepended)
constexpr int kH  = 256;
constexpr int kV  = 256;
constexpr int kUPAD = 112;  // 7 * 16
constexpr int kLDST = 264;  // padded f16 row stride (528B -> bank-conflict-free)

typedef _Float16 v16h __attribute__((ext_vector_type(16)));
typedef _Float16 v8h  __attribute__((ext_vector_type(8)));
typedef float    v8f  __attribute__((ext_vector_type(8)));

// ---------------------------------------------------------------------------
// f32 -> f16 conversion (Wj staging for WMMA B-matrix)
// ---------------------------------------------------------------------------
__global__ void cvt_f16_kernel(const float* __restrict__ in,
                               _Float16* __restrict__ out, int n) {
  int i = blockIdx.x * blockDim.x + threadIdx.x;
  if (i < n) out[i] = (_Float16)in[i];
}

// ---------------------------------------------------------------------------
// Generic row GEMM: out[row][h] = bias[h] + dot(x[row][:], W[h][:]),  K=N=256
// Used for enc_proj (x = encoder_output, W = We) and pred_proj (x = h1, W = Wp)
// ---------------------------------------------------------------------------
__global__ void __launch_bounds__(256)
rowgemm256_kernel(const float* __restrict__ x, const float* __restrict__ W,
                  const float* __restrict__ bias, float* __restrict__ out) {
  __shared__ __align__(16) float xs[kH];
  const int row = blockIdx.x;
  const int h   = threadIdx.x;
  xs[h] = x[(size_t)row * kH + h];
  __syncthreads();
  const float4* w4 = reinterpret_cast<const float4*>(W + (size_t)h * kH);
  const float4* x4 = reinterpret_cast<const float4*>(xs);
  float acc = bias[h];
#pragma unroll 4
  for (int k = 0; k < kH / 4; ++k) {
    float4 wv = w4[k];
    float4 xv = x4[k];
    acc += wv.x * xv.x + wv.y * xv.y + wv.z * xv.z + wv.w * xv.w;
  }
  out[(size_t)row * kH + h] = acc;
}

// ---------------------------------------------------------------------------
// LSTM input-precompute: pre[row][g] = b_ih[g] + b_hh[g] + dot(xrow, w_ih[g][:])
// row = b*kU1 + u.  If tgt != nullptr, xrow = emb[prepend-blank(targets)[b][u]],
// else xrow = xbase[row].
// Grid: (kB*kU1*4) blocks x 256 threads; each block does one quarter of 4H.
// ---------------------------------------------------------------------------
__global__ void __launch_bounds__(256)
pre_gemm_kernel(const float* __restrict__ xbase, const int* __restrict__ tgt,
                const float* __restrict__ emb, const float* __restrict__ w_ih,
                const float* __restrict__ b_ih, const float* __restrict__ b_hh,
                float* __restrict__ pre) {
  __shared__ __align__(16) float xs[kH];
  const int blk = blockIdx.x;
  const int row = blk >> 2;        // b*kU1 + u
  const int q   = blk & 3;
  const int tid = threadIdx.x;
  const int b = row / kU1;
  const int u = row % kU1;

  const float* xr;
  if (tgt != nullptr) {
    int id = (u == 0) ? 0 : tgt[b * kU + (u - 1)];
    xr = emb + (size_t)id * kH;
  } else {
    xr = xbase + (size_t)row * kH;
  }
  xs[tid] = xr[tid];
  __syncthreads();

  const int g = q * kH + tid;
  const float4* w4 = reinterpret_cast<const float4*>(w_ih + (size_t)g * kH);
  const float4* x4 = reinterpret_cast<const float4*>(xs);
  float acc = b_ih[g] + b_hh[g];
#pragma unroll 4
  for (int k = 0; k < kH / 4; ++k) {
    float4 wv = w4[k];
    float4 xv = x4[k];
    acc += wv.x * xv.x + wv.y * xv.y + wv.z * xv.z + wv.w * xv.w;
  }
  pre[(size_t)row * (4 * kH) + g] = acc;
}

// ---------------------------------------------------------------------------
// LSTM layer: one workgroup, 1024 threads, sequential over kU1 steps.
// Thread (q, j): gate unit g = q*256+j for all 4 batch rows.
// h kept transposed in LDS as h_t[k][b] so each k is one ds_load_b128.
// c kept in registers of the q==0 threads.
// ---------------------------------------------------------------------------
__global__ void __launch_bounds__(1024)
lstm_kernel(const float* __restrict__ pre, const float* __restrict__ w_hh,
            float* __restrict__ hout) {
  __shared__ __align__(16) float h_t[kH][4];
  __shared__ float gates[4][4][kH];   // [gate i,f,g,o][batch][j]
  const int tid = threadIdx.x;
  const int q = tid >> 8;
  const int j = tid & 255;
  float c0 = 0.f, c1 = 0.f, c2 = 0.f, c3 = 0.f;
  if (tid < kH) {
    h_t[tid][0] = 0.f; h_t[tid][1] = 0.f; h_t[tid][2] = 0.f; h_t[tid][3] = 0.f;
  }
  __syncthreads();

  const int g = q * kH + j;
  const float* wrow = w_hh + (size_t)g * kH;

  for (int s = 0; s < kU1; ++s) {
    float a0 = pre[((size_t)(0 * kU1 + s)) * (4 * kH) + g];
    float a1 = pre[((size_t)(1 * kU1 + s)) * (4 * kH) + g];
    float a2 = pre[((size_t)(2 * kU1 + s)) * (4 * kH) + g];
    float a3 = pre[((size_t)(3 * kU1 + s)) * (4 * kH) + g];
#pragma unroll 4
    for (int k = 0; k < kH; ++k) {
      float w = wrow[k];
      float4 hv = *reinterpret_cast<const float4*>(&h_t[k][0]);
      a0 += hv.x * w; a1 += hv.y * w; a2 += hv.z * w; a3 += hv.w * w;
    }
    float r0, r1, r2, r3;
    if (q == 2) {  // g gate: tanh
      r0 = tanhf(a0); r1 = tanhf(a1); r2 = tanhf(a2); r3 = tanhf(a3);
    } else {       // i, f, o: sigmoid
      r0 = 1.f / (1.f + expf(-a0));
      r1 = 1.f / (1.f + expf(-a1));
      r2 = 1.f / (1.f + expf(-a2));
      r3 = 1.f / (1.f + expf(-a3));
    }
    gates[q][0][j] = r0; gates[q][1][j] = r1;
    gates[q][2][j] = r2; gates[q][3][j] = r3;
    __syncthreads();

    if (q == 0) {
      float i0 = gates[0][0][j], f0 = gates[1][0][j], g0 = gates[2][0][j], o0 = gates[3][0][j];
      float i1 = gates[0][1][j], f1 = gates[1][1][j], g1 = gates[2][1][j], o1 = gates[3][1][j];
      float i2 = gates[0][2][j], f2 = gates[1][2][j], g2 = gates[2][2][j], o2 = gates[3][2][j];
      float i3 = gates[0][3][j], f3 = gates[1][3][j], g3 = gates[2][3][j], o3 = gates[3][3][j];
      c0 = f0 * c0 + i0 * g0;  float hb0 = o0 * tanhf(c0);
      c1 = f1 * c1 + i1 * g1;  float hb1 = o1 * tanhf(c1);
      c2 = f2 * c2 + i2 * g2;  float hb2 = o2 * tanhf(c2);
      c3 = f3 * c3 + i3 * g3;  float hb3 = o3 * tanhf(c3);
      h_t[j][0] = hb0; h_t[j][1] = hb1; h_t[j][2] = hb2; h_t[j][3] = hb3;
      hout[((size_t)(0 * kU1 + s)) * kH + j] = hb0;
      hout[((size_t)(1 * kU1 + s)) * kH + j] = hb1;
      hout[((size_t)(2 * kU1 + s)) * kH + j] = hb2;
      hout[((size_t)(3 * kU1 + s)) * kH + j] = hb3;
    }
    __syncthreads();
  }
}

// ---------------------------------------------------------------------------
// Joint kernel: per (b,t) workgroup of 256 threads (8 waves).
//   1) build jA[u][k] = f16(tanh(enc[b,t,k] + pred[b,u,k])) in LDS (u padded
//      to 112 with zeros), row stride 264 halves for conflict-free b128 reads.
//   2) each wave owns N-tiles {w, w+8} of Wj^T and sweeps the 7 U-tiles,
//      accumulating with v_wmma_f32_16x16x32_f16 over 8 K-chunks.
// Fragment layouts per CDNA5 ISA 7.12.2 (16-bit A 16x32; 32-bit C/D 16x16).
// ---------------------------------------------------------------------------
__global__ void __launch_bounds__(256)
joint_kernel(const float* __restrict__ enc, const float* __restrict__ pred,
             const _Float16* __restrict__ Wjh, const float* __restrict__ bj,
             float* __restrict__ out) {
  __shared__ __align__(16) _Float16 jA[kUPAD * kLDST];
  const int row = blockIdx.x;          // b*kT + t
  const int b   = row / kT;
  const int tid = threadIdx.x;

  // ---- build phase: thread tid owns column k = tid ----
  {
    const float ev = enc[(size_t)row * kH + tid];
    const float* predb = pred + (size_t)b * kU1 * kH;
    for (int u = 0; u < kUPAD; ++u) {
      float v = 0.f;
      if (u < kU1) v = tanhf(ev + predb[(size_t)u * kH + tid]);
      jA[u * kLDST + tid] = (_Float16)v;
    }
  }
  __syncthreads();

  // ---- GEMM phase ----
  const int w   = tid >> 5;
  const int ln  = tid & 31;
  const int col = ln & 15;   // A: M row / B: N col / D: N col
  const int hi  = ln >> 4;   // K-half select for A/B, M-half select for D
  const int kh  = hi * 8;    // K-half offset in halves

  float* outp = out + (size_t)row * kU1 * kV;

  for (int nt = w; nt < 16; nt += 8) {
    const _Float16* wjp = Wjh + (size_t)(nt * 16 + col) * kH;
    if (nt + 8 < 16)
      __builtin_prefetch(Wjh + (size_t)((nt + 8) * 16 + col) * kH, 0, 1);

    v16h bf[8];
#pragma unroll
    for (int kc = 0; kc < 8; ++kc) {
      v8h lo = *reinterpret_cast<const v8h*>(wjp + kc * 32 + kh);
      v8h hi8 = *reinterpret_cast<const v8h*>(wjp + kc * 32 + 16 + kh);
      bf[kc] = __builtin_shufflevector(lo, hi8, 0, 1, 2, 3, 4, 5, 6, 7,
                                       8, 9, 10, 11, 12, 13, 14, 15);
    }
    const float bjv = bj[nt * 16 + col];

    for (int ut = 0; ut < 7; ++ut) {
      v8f c = {};
      const _Float16* ap = &jA[(ut * 16 + col) * kLDST + kh];
#pragma unroll
      for (int kc = 0; kc < 8; ++kc) {
        v8h alo = *reinterpret_cast<const v8h*>(ap + kc * 32);
        v8h ahi = *reinterpret_cast<const v8h*>(ap + kc * 32 + 16);
        v16h a = __builtin_shufflevector(alo, ahi, 0, 1, 2, 3, 4, 5, 6, 7,
                                         8, 9, 10, 11, 12, 13, 14, 15);
        c = __builtin_amdgcn_wmma_f32_16x16x32_f16(
            /*neg_a=*/false, a, /*neg_b=*/false, bf[kc],
            /*c_mod=*/(short)0, c, /*reuse_a=*/false, /*reuse_b=*/false);
      }
#pragma unroll
      for (int r = 0; r < 8; ++r) {
        int u = ut * 16 + r + 8 * hi;           // D: VGPR r -> M = r + 8*hi
        if (u < kU1)
          outp[(size_t)u * kV + nt * 16 + col] = c[r] + bjv;
      }
    }
  }
}

// ---------------------------------------------------------------------------
// Host-side launcher
// ---------------------------------------------------------------------------
extern "C" void kernel_launch(void* const* d_in, const int* in_sizes, int n_in,
                              void* d_out, int out_size, void* d_ws, size_t ws_size,
                              hipStream_t stream) {
  (void)in_sizes; (void)n_in; (void)out_size; (void)ws_size;

  const float* enc_out = (const float*)d_in[0];
  const int*   targets = (const int*)d_in[1];
  const float* emb     = (const float*)d_in[2];
  const float* w_ih0   = (const float*)d_in[3];
  const float* w_hh0   = (const float*)d_in[4];
  const float* b_ih0   = (const float*)d_in[5];
  const float* b_hh0   = (const float*)d_in[6];
  const float* w_ih1   = (const float*)d_in[7];
  const float* w_hh1   = (const float*)d_in[8];
  const float* b_ih1   = (const float*)d_in[9];
  const float* b_hh1   = (const float*)d_in[10];
  const float* We      = (const float*)d_in[11];
  const float* be      = (const float*)d_in[12];
  const float* Wp      = (const float*)d_in[13];
  const float* bp      = (const float*)d_in[14];
  const float* Wj      = (const float*)d_in[15];
  const float* bj      = (const float*)d_in[16];
  float* out = (float*)d_out;

  // Workspace layout (f32 units)
  float* ws     = (float*)d_ws;
  float* enc_ws = ws;                                   // kB*kT*kH   = 393216
  float* pre0   = enc_ws + (size_t)kB * kT * kH;        // kB*kU1*4H  = 397312
  float* h0     = pre0 + (size_t)kB * kU1 * 4 * kH;     // kB*kU1*kH  = 99328
  float* pre1   = h0 + (size_t)kB * kU1 * kH;           // 397312
  float* h1     = pre1 + (size_t)kB * kU1 * 4 * kH;     // 99328
  float* predp  = h1 + (size_t)kB * kU1 * kH;           // 99328
  _Float16* Wjh = (_Float16*)(predp + (size_t)kB * kU1 * kH);  // kV*kH halves

  // Stage Wj as f16 for WMMA B-matrix
  cvt_f16_kernel<<<(kV * kH + 255) / 256, 256, 0, stream>>>(Wj, Wjh, kV * kH);

  // Encoder projection: (B*T, H)
  rowgemm256_kernel<<<kB * kT, 256, 0, stream>>>(enc_out, We, be, enc_ws);

  // Prediction network
  pre_gemm_kernel<<<kB * kU1 * 4, 256, 0, stream>>>(nullptr, targets, emb,
                                                    w_ih0, b_ih0, b_hh0, pre0);
  lstm_kernel<<<1, 1024, 0, stream>>>(pre0, w_hh0, h0);
  pre_gemm_kernel<<<kB * kU1 * 4, 256, 0, stream>>>(h0, nullptr, nullptr,
                                                    w_ih1, b_ih1, b_hh1, pre1);
  lstm_kernel<<<1, 1024, 0, stream>>>(pre1, w_hh1, h1);
  rowgemm256_kernel<<<kB * kU1, 256, 0, stream>>>(h1, Wp, bp, predp);

  // Joint network: tanh + WMMA GEMM + bias, writes full (B,T,U+1,V) output
  joint_kernel<<<kB * kT, 256, 0, stream>>>(enc_ws, predp, Wjh, bj, out);
}
